// GDN_87368224735786
// MI455X (gfx1250) — matrix operand order repaired and checked by reading, not compiled, not run on hardware
//
#include <hip/hip_runtime.h>
#include <hip/hip_bf16.h>
#include <math.h>

// ---------------------------------------------------------------------------
// GDN forward for MI455X (gfx1250, wave32, WMMA 16x16x32 f16)
// B=64, N=512, K=32, D=64
// ---------------------------------------------------------------------------

#define Bn   64
#define Nn   512
#define Kk   32
#define Dd   64
#define ROWS (Bn * Nn)          // 32768
#define EPSv 1e-5f
#define SLOPE 0.2f

typedef __attribute__((ext_vector_type(16))) _Float16 v16h;
typedef __attribute__((ext_vector_type(8)))  float    v8f;

// ---------------------------------------------------------------------------
// Kernel A: per-node norm + emb-attention scalars  (512 blocks x 64 thr)
// ---------------------------------------------------------------------------
__global__ void gdn_node_stats(const float* __restrict__ emb,
                               const float* __restrict__ att_em_i,
                               const float* __restrict__ att_em_j,
                               float* __restrict__ nrm,
                               float* __restrict__ emi,
                               float* __restrict__ emj) {
    __shared__ float s0[Dd], s1[Dd], s2[Dd];
    const int n = blockIdx.x;
    const int d = threadIdx.x;
    float e = emb[n * Dd + d];
    s0[d] = e * e;
    s1[d] = e * att_em_i[d];
    s2[d] = e * att_em_j[d];
    __syncthreads();
    for (int off = Dd / 2; off > 0; off >>= 1) {
        if (d < off) { s0[d] += s0[d + off]; s1[d] += s1[d + off]; s2[d] += s2[d + off]; }
        __syncthreads();
    }
    if (d == 0) { nrm[n] = sqrtf(s0[0]); emi[n] = s1[0]; emj[n] = s2[0]; }
}

// ---------------------------------------------------------------------------
// A-fragment loader: 16-bit A 16x32 (MxK) layout per ISA 7.12.2
//   lane m = lane&15, kh = lane>>4
//   VGPR r<4 : K = 2r   + 8*kh ;  r>=4 : K = 16 + 2(r-4) + 8*kh
// ---------------------------------------------------------------------------
__device__ inline v16h load_afrag_f32(const float* __restrict__ row, int k0, int lane) {
    const int kh = lane >> 4;
    v16h a;
#pragma unroll
    for (int r = 0; r < 8; ++r) {
        int kb = ((r < 4) ? (2 * r) : (16 + 2 * (r - 4))) + 8 * kh + k0;
        a[2 * r]     = (_Float16)row[kb];
        a[2 * r + 1] = (_Float16)row[kb + 1];
    }
    return a;
}

// B-fragment loader: 16-bit B 32x16 (KxN); lane n = lane&15, ks = lane>>4
//   half (2r+h) of lane <-> K = (r<4 ? 2r : 16+2(r-4)) + 8*ks + h ; `row` points at the data
//   holding column (n0+n) with contiguous K (i.e. the row-major transpose src)
__device__ inline v16h load_bfrag_f32(const float* __restrict__ row, int k0, int lane) {
    const int ks = lane >> 4;
    v16h b;
#pragma unroll
    for (int r = 0; r < 8; ++r) {
        int k = ((r < 4) ? (2 * r) : (16 + 2 * (r - 4))) + 8 * ks + k0;
        b[2 * r]     = (_Float16)row[k];
        b[2 * r + 1] = (_Float16)row[k + 1];
    }
    return b;
}

// ---------------------------------------------------------------------------
// Kernel B1: Gram matrix G = emb @ emb.T   (512x512x64) via WMMA
//   1024 tiles of 16x16, one tile per wave; 128 blocks x 256 thr (8 waves)
// ---------------------------------------------------------------------------
__global__ void gdn_gram_wmma(const float* __restrict__ emb,
                              float* __restrict__ G) {
    const int lane = threadIdx.x & 31;
    const int wave = blockIdx.x * (blockDim.x >> 5) + (threadIdx.x >> 5);
    const int ti = wave >> 5;            // 0..31 tile row
    const int tj = wave & 31;            // 0..31 tile col

    const float* arow = emb + (ti * 16 + (lane & 15)) * Dd; // A: rows of emb
    const float* brow = emb + (tj * 16 + (lane & 15)) * Dd; // B[k][j] = emb[j][k]

    v16h a0 = load_afrag_f32(arow, 0, lane);
    v16h a1 = load_afrag_f32(arow, 32, lane);
    v16h b0 = load_bfrag_f32(brow, 0, lane);
    v16h b1 = load_bfrag_f32(brow, 32, lane);

    v8f c = {};
    c = __builtin_amdgcn_wmma_f32_16x16x32_f16(false, a0, false, b0, (short)0, c, false, false);
    c = __builtin_amdgcn_wmma_f32_16x16x32_f16(false, a1, false, b1, (short)0, c, false, false);

#pragma unroll
    for (int v = 0; v < 8; ++v) {
        int row = ti * 16 + v + 8 * (lane >> 4);
        int col = tj * 16 + (lane & 15);
        G[row * Nn + col] = c[v];
    }
}

// ---------------------------------------------------------------------------
// Kernel B2: per-row cosine top-32 (iterative argmax, lowest-index tiebreak)
//   512 blocks x 256 thr
// ---------------------------------------------------------------------------
__global__ void gdn_topk(const float* __restrict__ G,
                         const float* __restrict__ nrm,
                         int* __restrict__ topk) {
    __shared__ float vals[Nn];
    __shared__ float rv[256];
    __shared__ int   ri[256];
    const int i = blockIdx.x;
    const int t = threadIdx.x;
    const float ni = nrm[i];
    for (int j = t; j < Nn; j += 256)
        vals[j] = G[i * Nn + j] / (ni * nrm[j]);
    __syncthreads();

    for (int it = 0; it < Kk; ++it) {
        float v0 = vals[t], v1 = vals[t + 256];
        float bv; int bi;
        if (v0 >= v1) { bv = v0; bi = t; } else { bv = v1; bi = t + 256; }
        rv[t] = bv; ri[t] = bi;
        __syncthreads();
        for (int off = 128; off > 0; off >>= 1) {
            if (t < off) {
                float ov = rv[t + off]; int oi = ri[t + off];
                if (ov > rv[t] || (ov == rv[t] && oi < ri[t])) { rv[t] = ov; ri[t] = oi; }
            }
            __syncthreads();
        }
        if (t == 0) { topk[i * Kk + it] = ri[0]; vals[ri[0]] = -3.0e38f; }
        __syncthreads();
    }
}

// ---------------------------------------------------------------------------
// Kernel C: xl = x @ lin_W.T   (32768 x 64 x 64) via WMMA
//   8192 tiles; 1024 blocks x 256 thr (8 waves, one tile each)
// ---------------------------------------------------------------------------
__global__ void gdn_xl_wmma(const float* __restrict__ x,
                            const float* __restrict__ lin_W,
                            float* __restrict__ xl) {
    const int lane = threadIdx.x & 31;
    const int wave = blockIdx.x * (blockDim.x >> 5) + (threadIdx.x >> 5);
    const int tm = wave >> 2;            // 0..2047
    const int tn = wave & 3;             // 0..3

    const float* arow = x + (tm * 16 + (lane & 15)) * Dd;          // rows of x
    const float* brow = lin_W + (tn * 16 + (lane & 15)) * Dd;      // B[k][n]=lin_W[n][k]

    v16h a0 = load_afrag_f32(arow, 0, lane);
    v16h a1 = load_afrag_f32(arow, 32, lane);
    v16h b0 = load_bfrag_f32(brow, 0, lane);
    v16h b1 = load_bfrag_f32(brow, 32, lane);

    v8f c = {};
    c = __builtin_amdgcn_wmma_f32_16x16x32_f16(false, a0, false, b0, (short)0, c, false, false);
    c = __builtin_amdgcn_wmma_f32_16x16x32_f16(false, a1, false, b1, (short)0, c, false, false);

#pragma unroll
    for (int v = 0; v < 8; ++v) {
        int row = tm * 16 + v + 8 * (lane >> 4);
        int col = tn * 16 + (lane & 15);
        xl[row * Dd + col] = c[v];
    }
}

// ---------------------------------------------------------------------------
// Kernel D: per-row attention scalars ai = xl.att_i, aj = xl.att_j
//   128 blocks x 256 thr, one row per thread
// ---------------------------------------------------------------------------
__global__ void gdn_row_scalars(const float* __restrict__ xl,
                                const float* __restrict__ att_i,
                                const float* __restrict__ att_j,
                                float* __restrict__ ai,
                                float* __restrict__ aj) {
    const int m = blockIdx.x * blockDim.x + threadIdx.x;
    if (m >= ROWS) return;
    const float* r = xl + m * Dd;
    float si = 0.f, sj = 0.f;
#pragma unroll 8
    for (int k = 0; k < Dd; ++k) { float v = r[k]; si += v * att_i[k]; sj += v * att_j[k]; }
    ai[m] = si; aj[m] = sj;
}

// ---------------------------------------------------------------------------
// Kernel E: fused per-node softmax + aggregation + BN/relu/emb/BN/relu + out
//   32768 blocks x 64 thr (one block per (b,i) node)
// ---------------------------------------------------------------------------
__global__ void gdn_node_fused(const float* __restrict__ xl,
                               const float* __restrict__ emb,
                               const int*   __restrict__ topk,
                               const float* __restrict__ ai,
                               const float* __restrict__ aj,
                               const float* __restrict__ emi,
                               const float* __restrict__ emj,
                               const float* __restrict__ gnn_bias,
                               const float* __restrict__ g1, const float* __restrict__ b1,
                               const float* __restrict__ m1, const float* __restrict__ v1,
                               const float* __restrict__ go, const float* __restrict__ bo,
                               const float* __restrict__ mo, const float* __restrict__ vo,
                               const float* __restrict__ out_W,
                               const float* __restrict__ out_b,
                               float* __restrict__ out) {
    const int g = blockIdx.x;            // global node 0..32767
    const int b = g >> 9;                // batch
    const int i = g & (Nn - 1);          // node in graph
    const int t = threadIdx.x;           // 0..63

    __shared__ float alpha[Kk + 1];
    __shared__ int   srcs[Kk + 1];
    __shared__ float red[Dd];

    if (t < Kk + 1) {
        int sl = (t < Kk) ? topk[i * Kk + t] : i;  // last slot = self loop
        bool valid = (t == Kk) || (sl != i);       // remove_self_loops on topk edges
        float a = ai[g] + emi[i] + aj[b * Nn + sl] + emj[sl];
        a = (a >= 0.f) ? a : SLOPE * a;            // leaky relu
        alpha[t] = valid ? a : -1e9f;
        srcs[t] = sl;
    }
    __syncthreads();

    if (t == 0) {                                  // 33-wide softmax, serial is cheap
        float mx = alpha[0];
        for (int e = 1; e <= Kk; ++e) mx = fmaxf(mx, alpha[e]);
        float den = 0.f;
        for (int e = 0; e <= Kk; ++e) {
            float x = (alpha[e] <= -1e8f) ? 0.f : expf(alpha[e] - mx);
            alpha[e] = x; den += x;
        }
        float inv = 1.f / den;
        for (int e = 0; e <= Kk; ++e) alpha[e] *= inv;
    }
    __syncthreads();

    // channel t: gather-weighted sum over the 33 edges
    float acc = 0.f;
#pragma unroll 8
    for (int e = 0; e <= Kk; ++e)
        acc += alpha[e] * xl[(b * Nn + srcs[e]) * Dd + t];

    float h = acc + gnn_bias[t];
    h = (h - m1[t]) * rsqrtf(v1[t] + EPSv) * g1[t] + b1[t];   // BN1 (eval)
    h = fmaxf(h, 0.f);                                        // relu
    h = h * emb[i * Dd + t];                                  // * node embedding
    h = (h - mo[t]) * rsqrtf(vo[t] + EPSv) * go[t] + bo[t];   // BN out (eval)
    h = fmaxf(h, 0.f);                                        // relu
    red[t] = h * out_W[t];                                    // Linear(dim,1)
    __syncthreads();
    for (int off = Dd / 2; off > 0; off >>= 1) {
        if (t < off) red[t] += red[t + off];
        __syncthreads();
    }
    if (t == 0) out[g] = red[0] + out_b[0];
}

// ---------------------------------------------------------------------------
// Host-side launcher
// ---------------------------------------------------------------------------
extern "C" void kernel_launch(void* const* d_in, const int* in_sizes, int n_in,
                              void* d_out, int out_size, void* d_ws, size_t ws_size,
                              hipStream_t stream) {
    const float* data      = (const float*)d_in[0];
    // d_in[1] = org_edge_index (unused by GDN.forward)
    const float* emb       = (const float*)d_in[2];
    const float* lin_W     = (const float*)d_in[3];
    const float* att_i     = (const float*)d_in[4];
    const float* att_j     = (const float*)d_in[5];
    const float* att_em_i  = (const float*)d_in[6];
    const float* att_em_j  = (const float*)d_in[7];
    const float* gnn_bias  = (const float*)d_in[8];
    const float* bn1_gamma = (const float*)d_in[9];
    const float* bn1_beta  = (const float*)d_in[10];
    const float* bn1_mean  = (const float*)d_in[11];
    const float* bn1_var   = (const float*)d_in[12];
    const float* bno_gamma = (const float*)d_in[13];
    const float* bno_beta  = (const float*)d_in[14];
    const float* bno_mean  = (const float*)d_in[15];
    const float* bno_var   = (const float*)d_in[16];
    const float* out_W     = (const float*)d_in[17];
    const float* out_b     = (const float*)d_in[18];
    float*       out       = (float*)d_out;

    // workspace layout (256B aligned chunks)
    char* ws = (char*)d_ws;
    size_t off = 0;
    auto alloc = [&](size_t bytes) { char* p = ws + off; off = (off + bytes + 255) & ~size_t(255); return p; };
    float* G    = (float*)alloc(size_t(Nn) * Nn * sizeof(float));      // 1 MB
    float* nrm  = (float*)alloc(Nn * sizeof(float));
    float* emi  = (float*)alloc(Nn * sizeof(float));
    float* emj  = (float*)alloc(Nn * sizeof(float));
    int*   topk = (int*)  alloc(Nn * Kk * sizeof(int));                // 64 KB
    float* xl   = (float*)alloc(size_t(ROWS) * Dd * sizeof(float));    // 8 MB
    float* ai   = (float*)alloc(ROWS * sizeof(float));
    float* aj   = (float*)alloc(ROWS * sizeof(float));
    (void)ws_size; (void)in_sizes; (void)n_in; (void)out_size;

    // A: node norms + emb scalars
    gdn_node_stats<<<Nn, Dd, 0, stream>>>(emb, att_em_i, att_em_j, nrm, emi, emj);
    // B1: gram matrix via WMMA (1024 tiles / 8 waves per block)
    gdn_gram_wmma<<<128, 256, 0, stream>>>(emb, G);
    // C: xl GEMM via WMMA (8192 tiles / 8 waves per block) — independent of B
    gdn_xl_wmma<<<1024, 256, 0, stream>>>(data, lin_W, xl);
    // B2: top-32 per row
    gdn_topk<<<Nn, 256, 0, stream>>>(G, nrm, topk);
    // D: per-row attention scalars
    gdn_row_scalars<<<ROWS / 256, 256, 0, stream>>>(xl, att_i, att_j, ai, aj);
    // E: fused softmax + aggregation + BN/relu/out per node
    gdn_node_fused<<<ROWS, Dd, 0, stream>>>(xl, emb, topk, ai, aj, emi, emj,
                                            gnn_bias,
                                            bn1_gamma, bn1_beta, bn1_mean, bn1_var,
                                            bno_gamma, bno_beta, bno_mean, bno_var,
                                            out_W, out_b, out);
}
